// AttnDecoderRNN_28845000360408
// MI455X (gfx1250) — matrix operand order repaired and checked by reading, not compiled, not run on hardware
//
#include <hip/hip_runtime.h>
#include <math.h>

// ---------------- problem constants ----------------
#define B_  32
#define T_  20
#define L_  49
#define F_  512
#define H_  1024
#define V_  32000

typedef __bf16 bf16;
typedef __attribute__((ext_vector_type(16))) __bf16 v16bf;
typedef __attribute__((ext_vector_type(8)))  float  v8f;

union Frag16 { v16bf v; uint4 q[2]; };

__device__ __forceinline__ bf16 f2b(float x) {
    union { float f; unsigned u; } v; v.f = x;
    unsigned r = v.u + 0x7FFFu + ((v.u >> 16) & 1u);   // round-to-nearest-even
    unsigned short h = (unsigned short)(r >> 16);
    bf16 o; __builtin_memcpy(&o, &h, 2);
    return o;
}
__device__ __forceinline__ float sigmoidf_(float x) { return 1.0f / (1.0f + __expf(-x)); }

// ---------------- WMMA fragment loaders (ISA 7.12.2 bf16 layouts) ----------------
// A 16x32 (MxK): lane m = lane&15, g = lane>>4.
//   elems 0..7  -> K = kbase + g*8 + e        (contiguous 16B)
//   elems 8..15 -> K = kbase + 16 + g*8 + e'  (contiguous 16B)
__device__ __forceinline__ v16bf load_a(const bf16* __restrict__ A, int lda,
                                        int mbase, int kbase, int lane) {
    int m = mbase + (lane & 15);
    int g = lane >> 4;
    const bf16* p = A + (size_t)m * lda + kbase + g * 8;
    Frag16 f;
    f.q[0] = *(const uint4*)(p);
    f.q[1] = *(const uint4*)(p + 16);
    return f.v;
}
// B 32x16 (KxN) sourced from row-major W (N,K): lane n = lane&15, g = lane>>4.
//   elems 0..7 -> K = kbase + g*8 + e, elems 8..15 -> K = kbase + 16 + g*8 + e'
__device__ __forceinline__ v16bf load_b(const bf16* __restrict__ W, int ldw,
                                        int nbase, int kbase, int lane) {
    int n = nbase + (lane & 15);
    int g = lane >> 4;
    const bf16* p = W + (size_t)n * ldw + kbase + g * 8;
    Frag16 f;
    f.q[0] = *(const uint4*)(p);
    f.q[1] = *(const uint4*)(p + 16);
    return f.v;
}

// ---------------- generic skinny GEMM: D(M,N) = A(M,K) @ W(N,K)^T + bias ----------------
// block = 128 threads (4 waves). wave w -> N strip [blockIdx.x*64 + 16w).
// grid.y -> 32-row M strips; each wave computes two 16x16 accumulators.
// ACT: 0 = none, 1 = tanh(d + extra[m,n]). OBF16: store bf16 instead of f32.
template <int ACT, bool OBF16>
__global__ void gemm_wmma(const bf16* __restrict__ A, const bf16* __restrict__ W,
                          const float* __restrict__ bias, const float* __restrict__ extra,
                          int ldExtra, void* __restrict__ Dp, int ldD, int K) {
    const int lane  = threadIdx.x & 31;
    const int wave  = threadIdx.x >> 5;
    const int nbase = blockIdx.x * 64 + wave * 16;
    const int mbase = blockIdx.y * 32;
    const bf16* Arow = A + (size_t)mbase * K;

    v8f acc0 = {}; v8f acc1 = {};
    for (int k0 = 0; k0 < K; k0 += 32) {
        v16bf bfr = load_b(W, K, nbase, k0, lane);
        v16bf a0  = load_a(Arow, K, 0,  k0, lane);
        v16bf a1  = load_a(Arow, K, 16, k0, lane);
        acc0 = __builtin_amdgcn_wmma_f32_16x16x32_bf16(false, a0, false, bfr,
                                                       (short)0, acc0, false, false);
        acc1 = __builtin_amdgcn_wmma_f32_16x16x32_bf16(false, a1, false, bfr,
                                                       (short)0, acc1, false, false);
    }

    const int n    = nbase + (lane & 15);
    const int half = lane >> 4;
    const float bs = bias ? bias[n] : 0.0f;
#pragma unroll
    for (int r = 0; r < 8; ++r) {
        int m0 = mbase + half * 8 + r;       // rows 0..15 of strip
        int m1 = m0 + 16;                    // rows 16..31 of strip
        float d0 = acc0[r] + bs;
        float d1 = acc1[r] + bs;
        if (ACT == 1) {
            d0 = tanhf(d0 + extra[(size_t)m0 * ldExtra + n]);
            d1 = tanhf(d1 + extra[(size_t)m1 * ldExtra + n]);
        }
        if (OBF16) {
            ((bf16*)Dp)[(size_t)m0 * ldD + n] = f2b(d0);
            ((bf16*)Dp)[(size_t)m1 * ldD + n] = f2b(d1);
        } else {
            ((float*)Dp)[(size_t)m0 * ldD + n] = d0;
            ((float*)Dp)[(size_t)m1 * ldD + n] = d1;
        }
    }
}

// ---------------- prep kernels ----------------
__global__ void pack_wcat(const float* __restrict__ Wih, const float* __restrict__ Whh,
                          bf16* __restrict__ out) {          // (4H, 2H)
    size_t idx = (size_t)blockIdx.x * blockDim.x + threadIdx.x;
    if (idx >= (size_t)4 * H_ * 2 * H_) return;
    int n = (int)(idx / (2 * H_)), k = (int)(idx % (2 * H_));
    float v = (k < H_) ? Wih[(size_t)n * H_ + k] : Whh[(size_t)n * H_ + (k - H_)];
    out[idx] = f2b(v);
}
__global__ void pack_wdec(const float* __restrict__ Wh2o, const float* __restrict__ Wc2o,
                          bf16* __restrict__ out) {          // (F, H+F)
    size_t idx = (size_t)blockIdx.x * blockDim.x + threadIdx.x;
    if (idx >= (size_t)F_ * (H_ + F_)) return;
    int n = (int)(idx / (H_ + F_)), k = (int)(idx % (H_ + F_));
    float v = (k < H_) ? Wh2o[(size_t)n * H_ + k] : Wc2o[(size_t)n * F_ + (k - H_)];
    out[idx] = f2b(v);
}
__global__ void f2bf_kernel(const float* __restrict__ in, bf16* __restrict__ out, size_t n) {
    size_t idx = (size_t)blockIdx.x * blockDim.x + threadIdx.x;
    if (idx < n) out[idx] = f2b(in[idx]);
}
__global__ void add_bias_kernel(const float* __restrict__ a, const float* __restrict__ b,
                                float* __restrict__ out, int n) {
    int idx = blockIdx.x * blockDim.x + threadIdx.x;
    if (idx < n) out[idx] = a[idx] + b[idx];
}
__global__ void featsT_kernel(const float* __restrict__ features, bf16* __restrict__ out) {
    // features (B,F,L) -> out (B*L, F) bf16
    size_t idx = (size_t)blockIdx.x * blockDim.x + threadIdx.x;
    if (idx >= (size_t)B_ * L_ * F_) return;
    int row = (int)(idx / F_), f = (int)(idx % F_);
    int b = row / L_, l = row % L_;
    out[idx] = f2b(features[((size_t)b * F_ + f) * L_ + l]);
}
__global__ void iw_kernel(const float* __restrict__ embed, const int* __restrict__ captions,
                          float* __restrict__ iw_all) {      // (B,T,F)
    size_t idx = (size_t)blockIdx.x * blockDim.x + threadIdx.x;
    if (idx >= (size_t)B_ * T_ * F_) return;
    int b = (int)(idx / (T_ * F_));
    int r = (int)(idx % (T_ * F_));
    int t = r / F_, f = r % F_;
    float v = 0.0f;
    if (t > 0) v = embed[(size_t)captions[b * T_ + (t - 1)] * F_ + f];
    iw_all[idx] = v;
}
__global__ void init_h_kernel(const float* __restrict__ features,
                              const float* __restrict__ W_init, const float* __restrict__ b_init,
                              float* __restrict__ h, float* __restrict__ c) {
    __shared__ float fm[F_];
    int b = blockIdx.x, tid = threadIdx.x;
    for (int f = tid; f < F_; f += blockDim.x) {
        float s = 0.0f;
        const float* p = features + ((size_t)b * F_ + f) * L_;
        for (int l = 0; l < L_; ++l) s += p[l];
        fm[f] = s * (1.0f / (float)L_);
    }
    __syncthreads();
    for (int j = tid; j < H_; j += blockDim.x) {
        float d = b_init[j];
        const float* w = W_init + (size_t)j * F_;
        for (int k = 0; k < F_; ++k) d += fm[k] * w[k];
        float v = tanhf(d);
        h[(size_t)b * H_ + j] = v;
        c[(size_t)b * H_ + j] = v;
    }
}

// ---------------- per-step kernels ----------------
__global__ void attn_kernel(const float* __restrict__ feats_proj,  // (B,L,H)
                            const float* __restrict__ features,    // (B,F,L)
                            const float* __restrict__ h,           // (B,H)
                            const float* __restrict__ iw_all,      // (B,T,F)
                            int t,
                            float* __restrict__ ctx,               // (B,F)
                            bf16* __restrict__ Xg,                 // (B,2H) = [ctx|iw|h]
                            bf16* __restrict__ Xd) {               // (B,H+F), writes ctx cols
    __shared__ float sc[L_];
    __shared__ float al[L_];
    int b = blockIdx.x, tid = threadIdx.x;
    if (tid < L_) sc[tid] = 0.0f;
    __syncthreads();

    // scores[l] = h[b,:] . feats_proj[b,l,:]  (32 segments of 32 per l)
    const float* hb = h + (size_t)b * H_;
    for (int i = tid; i < L_ * 32; i += blockDim.x) {
        int l = i >> 5, seg = i & 31;
        const float* fp = feats_proj + ((size_t)b * L_ + l) * H_ + seg * 32;
        const float* hp = hb + seg * 32;
        float s = 0.0f;
#pragma unroll
        for (int k = 0; k < 32; ++k) s += hp[k] * fp[k];
        atomicAdd(&sc[l], s);
    }
    __syncthreads();

    float mx = -1e30f;
    for (int l = 0; l < L_; ++l) mx = fmaxf(mx, sc[l]);
    if (tid < L_) al[tid] = __expf(sc[tid] - mx);
    __syncthreads();
    float ssum = 0.0f;
    for (int l = 0; l < L_; ++l) ssum += al[l];
    float inv = 1.0f / ssum;

    // ctx[f] = sum_l alpha[l] * features[b,f,l]
    for (int f = tid; f < F_; f += blockDim.x) {
        const float* p = features + ((size_t)b * F_ + f) * L_;
        float s = 0.0f;
        for (int l = 0; l < L_; ++l) s += al[l] * p[l];
        s *= inv;
        ctx[(size_t)b * F_ + f] = s;
        Xg[(size_t)b * 2 * H_ + f]       = f2b(s);   // x = [ctx, iw]
        Xd[(size_t)b * (H_ + F_) + H_ + f] = f2b(s); // dec input tail
    }
    // input word (bf16) into Xg cols [F, 2F)
    const float* iw = iw_all + ((size_t)b * T_ + t) * F_;
    for (int f = tid; f < F_; f += blockDim.x)
        Xg[(size_t)b * 2 * H_ + F_ + f] = f2b(iw[f]);
    // h (bf16) into Xg cols [H, 2H)
    for (int j = tid; j < H_; j += blockDim.x)
        Xg[(size_t)b * 2 * H_ + H_ + j] = f2b(hb[j]);
}

__global__ void lstm_kernel(const float* __restrict__ gates,  // (B,4H) order i,f,g,o
                            float* __restrict__ h, float* __restrict__ c,
                            bf16* __restrict__ XdH) {          // (B,H+F) head cols
    int idx = blockIdx.x * blockDim.x + threadIdx.x;
    if (idx >= B_ * H_) return;
    int b = idx / H_, j = idx % H_;
    const float* g = gates + (size_t)b * 4 * H_;
    float i_ = sigmoidf_(g[j]);
    float f_ = sigmoidf_(g[H_ + j]);
    float g_ = tanhf(g[2 * H_ + j]);
    float o_ = sigmoidf_(g[3 * H_ + j]);
    float cn = f_ * c[idx] + i_ * g_;
    float hn = o_ * tanhf(cn);
    c[idx] = cn;
    h[idx] = hn;
    XdH[(size_t)b * (H_ + F_) + j] = f2b(hn);
}

// ---------------- host launch ----------------
static inline size_t rup(size_t x) { return (x + 255) & ~(size_t)255; }

extern "C" void kernel_launch(void* const* d_in, const int* in_sizes, int n_in,
                              void* d_out, int out_size, void* d_ws, size_t ws_size,
                              hipStream_t stream) {
    (void)in_sizes; (void)n_in; (void)out_size; (void)ws_size;
    const float* features = (const float*)d_in[0];
    const int*   captions = (const int*)d_in[1];
    // d_in[2] lengths unused
    const float* embed   = (const float*)d_in[3];
    const float* Wa      = (const float*)d_in[4];
    const float* ba      = (const float*)d_in[5];
    const float* W_init  = (const float*)d_in[6];
    const float* b_init  = (const float*)d_in[7];
    const float* W_ih    = (const float*)d_in[8];
    const float* W_hh    = (const float*)d_in[9];
    const float* b_ih    = (const float*)d_in[10];
    const float* b_hh    = (const float*)d_in[11];
    const float* W_c2o   = (const float*)d_in[12];
    const float* b_c2o   = (const float*)d_in[13];
    const float* W_h2o   = (const float*)d_in[14];
    const float* b_h2o   = (const float*)d_in[15];
    const float* W_out   = (const float*)d_in[16];
    const float* b_out   = (const float*)d_in[17];
    float* out = (float*)d_out;

    // workspace bump allocator (~60 MB total)
    char* wp = (char*)d_ws;
    auto alloc = [&](size_t bytes) { void* p = wp; wp += rup(bytes); return p; };
    bf16*  W_cat   = (bf16*)alloc((size_t)4 * H_ * 2 * H_ * 2);
    bf16*  W_dec   = (bf16*)alloc((size_t)F_ * (H_ + F_) * 2);
    bf16*  W_outb  = (bf16*)alloc((size_t)V_ * F_ * 2);
    bf16*  Wab     = (bf16*)alloc((size_t)H_ * F_ * 2);
    bf16*  featsT  = (bf16*)alloc((size_t)B_ * L_ * F_ * 2);
    float* fproj   = (float*)alloc((size_t)B_ * L_ * H_ * 4);
    float* bias_g  = (float*)alloc((size_t)4 * H_ * 4);
    float* bias_d  = (float*)alloc((size_t)F_ * 4);
    float* iw_all  = (float*)alloc((size_t)B_ * T_ * F_ * 4);
    float* hbuf    = (float*)alloc((size_t)B_ * H_ * 4);
    float* cbuf    = (float*)alloc((size_t)B_ * H_ * 4);
    float* ctx     = (float*)alloc((size_t)B_ * F_ * 4);
    bf16*  Xg      = (bf16*)alloc((size_t)B_ * 2 * H_ * 2);
    bf16*  Xd      = (bf16*)alloc((size_t)B_ * (H_ + F_) * 2);
    bf16*  Xo      = (bf16*)alloc((size_t)B_ * F_ * 2);
    float* gates   = (float*)alloc((size_t)B_ * 4 * H_ * 4);

    const int TB = 256;
    // --- prep ---
    pack_wcat<<<(4 * H_ * 2 * H_ + TB - 1) / TB, TB, 0, stream>>>(W_ih, W_hh, W_cat);
    pack_wdec<<<(F_ * (H_ + F_) + TB - 1) / TB, TB, 0, stream>>>(W_h2o, W_c2o, W_dec);
    f2bf_kernel<<<((size_t)V_ * F_ + TB - 1) / TB, TB, 0, stream>>>(W_out, W_outb, (size_t)V_ * F_);
    f2bf_kernel<<<(H_ * F_ + TB - 1) / TB, TB, 0, stream>>>(Wa, Wab, (size_t)H_ * F_);
    add_bias_kernel<<<(4 * H_ + TB - 1) / TB, TB, 0, stream>>>(b_ih, b_hh, bias_g, 4 * H_);
    add_bias_kernel<<<(F_ + TB - 1) / TB, TB, 0, stream>>>(b_h2o, b_c2o, bias_d, F_);
    featsT_kernel<<<(B_ * L_ * F_ + TB - 1) / TB, TB, 0, stream>>>(features, featsT);
    iw_kernel<<<(B_ * T_ * F_ + TB - 1) / TB, TB, 0, stream>>>(embed, captions, iw_all);
    init_h_kernel<<<B_, TB, 0, stream>>>(features, W_init, b_init, hbuf, cbuf);
    // feats_proj (B*L=1568 rows = 49 strips of 32, N=H, K=F) via WMMA
    gemm_wmma<0, false><<<dim3(H_ / 64, (B_ * L_) / 32), 128, 0, stream>>>(
        featsT, Wab, ba, nullptr, 0, fproj, H_, F_);

    // --- timestep loop (sequential recurrence) ---
    for (int t = 0; t < T_; ++t) {
        attn_kernel<<<B_, TB, 0, stream>>>(fproj, features, hbuf, iw_all, t, ctx, Xg, Xd);
        // gates = [ctx|iw|h] @ [W_ih|W_hh]^T + (b_ih+b_hh)   (32 x 4096, K=2048)
        gemm_wmma<0, false><<<dim3(4 * H_ / 64, 1), 128, 0, stream>>>(
            Xg, W_cat, bias_g, nullptr, 0, gates, 4 * H_, 2 * H_);
        lstm_kernel<<<(B_ * H_ + TB - 1) / TB, TB, 0, stream>>>(gates, hbuf, cbuf, Xd);
        // dec = tanh([h|ctx] @ [W_h2o|W_c2o]^T + bias + iw_t)  -> bf16 (32 x 512, K=1536)
        gemm_wmma<1, true><<<dim3(F_ / 64, 1), 128, 0, stream>>>(
            Xd, W_dec, bias_d, iw_all + (size_t)t * F_, T_ * F_, Xo, F_, H_ + F_);
        // logits = dec @ W_out^T + b_out  -> d_out[b*T*V + t*V + v]  (32 x 32000, K=512)
        gemm_wmma<0, false><<<dim3(V_ / 64, 1), 128, 0, stream>>>(
            Xo, W_outb, b_out, nullptr, 0, out + (size_t)t * V_, T_ * V_, F_);
    }
}
